// PointNetV2_12790412607708
// MI455X (gfx1250) — matrix operand order repaired and checked by reading, not compiled, not run on hardware
//
#include <hip/hip_runtime.h>
#include <hip/hip_bf16.h>

typedef __bf16 bf16_t;
typedef __attribute__((ext_vector_type(16))) __bf16 bf16x16;
typedef __attribute__((ext_vector_type(8)))  __bf16 bf16x8;
typedef __attribute__((ext_vector_type(8)))  float  f32x8;

#define BN_EPS 1e-5f

// ---------------------------------------------------------------------------
// Layout transpose: xyz (B,3,N) -> pts (B,N,3)
// ---------------------------------------------------------------------------
__global__ void k_transpose(const float* __restrict__ in, float* __restrict__ out,
                            int Bc, int Npts) {
  int i = blockIdx.x * blockDim.x + threadIdx.x;
  int total = Bc * 3 * Npts;
  if (i >= total) return;
  int n = i % Npts;
  int c = (i / Npts) % 3;
  int b = i / (3 * Npts);
  out[((size_t)b * Npts + n) * 3 + c] = in[i];
}

// ---------------------------------------------------------------------------
// Farthest point sampling: one block (256 thr) per batch, sequential scan.
// Matches reference: fidx[it] = far BEFORE update; far starts at 0.
// ---------------------------------------------------------------------------
__global__ void k_fps(const float* __restrict__ pts, float* __restrict__ dist,
                      int* __restrict__ fidx, int Npts, int npoint) {
  int b = blockIdx.x;
  const float* p = pts + (size_t)b * Npts * 3;
  float* d = dist + (size_t)b * Npts;
  __shared__ float sv[256];
  __shared__ int   si[256];
  __shared__ int   sfar;
  int t = threadIdx.x;
  for (int i = t; i < Npts; i += 256) d[i] = 1e10f;
  if (t == 0) sfar = 0;
  __syncthreads();
  for (int it = 0; it < npoint; ++it) {
    int far = sfar;
    if (t == 0) fidx[b * npoint + it] = far;
    float cx = p[far * 3 + 0], cy = p[far * 3 + 1], cz = p[far * 3 + 2];
    float best = -1e30f;
    int bidx = 0;
    for (int i = t; i < Npts; i += 256) {
      float dx = p[i * 3 + 0] - cx;
      float dy = p[i * 3 + 1] - cy;
      float dz = p[i * 3 + 2] - cz;
      float dd = dx * dx + dy * dy + dz * dz;
      float m = fminf(d[i], dd);
      d[i] = m;
      if (m > best) { best = m; bidx = i; }
    }
    sv[t] = best; si[t] = bidx;
    __syncthreads();
    for (int off = 128; off > 0; off >>= 1) {
      if (t < off) {
        float ov = sv[t + off]; int oi = si[t + off];
        if (ov > sv[t] || (ov == sv[t] && oi < si[t])) { sv[t] = ov; si[t] = oi; }
      }
      __syncthreads();
    }
    if (t == 0) sfar = si[0];
    __syncthreads();
  }
}

// ---------------------------------------------------------------------------
// Gather centroids: out(B,S,3) = pts[fidx]
// ---------------------------------------------------------------------------
__global__ void k_gather3(const float* __restrict__ pts, const int* __restrict__ fidx,
                          float* __restrict__ out, int Npts, int S, int total) {
  int i = blockIdx.x * blockDim.x + threadIdx.x;
  if (i >= total) return;
  int c = i % 3;
  int s = (i / 3) % S;
  int b = i / (3 * S);
  out[i] = pts[((size_t)b * Npts + fidx[b * S + s]) * 3 + c];
}

// ---------------------------------------------------------------------------
// Ball query: one thread per (b,s). First K in-range indices (ascending),
// padded with the first hit (matches reference sort/where semantics).
// ---------------------------------------------------------------------------
__global__ void k_ball(const float* __restrict__ pts, const float* __restrict__ cent,
                       int* __restrict__ gidx, int Npts, int S, int K, float r2) {
  int q = blockIdx.x * blockDim.x + threadIdx.x;
  if (q >= 8 * S) return;
  int b = q / S;
  const float* p = pts + (size_t)b * Npts * 3;
  float cx = cent[(size_t)q * 3 + 0];
  float cy = cent[(size_t)q * 3 + 1];
  float cz = cent[(size_t)q * 3 + 2];
  int* out = gidx + (size_t)q * K;
  int cnt = 0;
  for (int i = 0; i < Npts && cnt < K; ++i) {
    float dx = p[i * 3 + 0] - cx;
    float dy = p[i * 3 + 1] - cy;
    float dz = p[i * 3 + 2] - cz;
    if (dx * dx + dy * dy + dz * dz <= r2) out[cnt++] = i;
  }
  int first = (cnt > 0) ? out[0] : 0;
  for (; cnt < K; ++cnt) out[cnt] = first;
}

// ---------------------------------------------------------------------------
// Grouping: build bf16 GEMM input rows (zero-padded to Cp).
// ---------------------------------------------------------------------------
__global__ void k_group_first(const float* __restrict__ pts, const float* __restrict__ cent,
                              const int* __restrict__ gidx, bf16_t* __restrict__ X,
                              int Npts, int S, int K, int Cp, int rows) {
  int r = blockIdx.x * blockDim.x + threadIdx.x;
  if (r >= rows) return;
  int k = r % K;
  int s = (r / K) % S;
  int b = r / (K * S);
  int idx = gidx[(size_t)(b * S + s) * K + k];
  const float* p = pts + ((size_t)b * Npts + idx) * 3;
  const float* c = cent + (size_t)(b * S + s) * 3;
  bf16_t* o = X + (size_t)r * Cp;
  o[0] = (bf16_t)(p[0] - c[0]);
  o[1] = (bf16_t)(p[1] - c[1]);
  o[2] = (bf16_t)(p[2] - c[2]);
  for (int j = 3; j < Cp; ++j) o[j] = (bf16_t)0.f;
}

__global__ void k_group_feat(const float* __restrict__ pts, const float* __restrict__ cent,
                             const float* __restrict__ feat, const int* __restrict__ gidx,
                             bf16_t* __restrict__ X, int Npts, int S, int K,
                             int Cf, int Cp, int rows) {
  int r = blockIdx.x * blockDim.x + threadIdx.x;
  if (r >= rows) return;
  int k = r % K;
  int s = (r / K) % S;
  int b = r / (K * S);
  int idx = gidx[(size_t)(b * S + s) * K + k];
  const float* f = feat + ((size_t)b * Npts + idx) * Cf;
  const float* p = pts + ((size_t)b * Npts + idx) * 3;
  const float* c = cent + (size_t)(b * S + s) * 3;
  bf16_t* o = X + (size_t)r * Cp;
  for (int j = 0; j < Cf; ++j) o[j] = (bf16_t)f[j];
  o[Cf + 0] = (bf16_t)(p[0] - c[0]);
  o[Cf + 1] = (bf16_t)(p[1] - c[1]);
  o[Cf + 2] = (bf16_t)(p[2] - c[2]);
  for (int j = Cf + 3; j < Cp; ++j) o[j] = (bf16_t)0.f;
}

__global__ void k_group_all(const float* __restrict__ xyz, const float* __restrict__ feat,
                            bf16_t* __restrict__ X, int K, int Cf, int Cp, int rows) {
  int r = blockIdx.x * blockDim.x + threadIdx.x;
  if (r >= rows) return;
  const float* p = xyz + (size_t)r * 3;
  const float* f = feat + (size_t)r * Cf;
  bf16_t* o = X + (size_t)r * Cp;
  o[0] = (bf16_t)p[0];
  o[1] = (bf16_t)p[1];
  o[2] = (bf16_t)p[2];
  for (int j = 0; j < Cf; ++j) o[3 + j] = (bf16_t)f[j];
  for (int j = 3 + Cf; j < Cp; ++j) o[j] = (bf16_t)0.f;
}

// ---------------------------------------------------------------------------
// Weight f32 (O,C) -> bf16 (O,Cp) zero-padded.
// ---------------------------------------------------------------------------
__global__ void k_convw(const float* __restrict__ W, bf16_t* __restrict__ Wb,
                        int C, int Cp, int total) {
  int i = blockIdx.x * blockDim.x + threadIdx.x;
  if (i >= total) return;
  int c = i % Cp;
  int o = i / Cp;
  Wb[i] = (c < C) ? (bf16_t)W[(size_t)o * C + c] : (bf16_t)0.f;
}

__global__ void k_zero(float* __restrict__ p, int n) {
  int i = blockIdx.x * blockDim.x + threadIdx.x;
  if (i < n) p[i] = 0.f;
}

// ---------------------------------------------------------------------------
// WMMA GEMM: D(M,O) = X(M,Cp) * Wb(O,Cp)^T + bias. One wave computes a
// 16 x (16*NT) strip: the A fragment is loaded once and reused across NT
// independent WMMAs (independent accumulators -> no D->A/B hazards, good
// co-execution). bf16 inputs, f32 accumulate; epilogue adds bias,
// accumulates BN stats (sum/sumsq per channel via f32 atomics), stores
// bf16 pre-BN activation. A and B both use contiguous k order (k = 16*half + s).
// ---------------------------------------------------------------------------
template <int NT>
__global__ __launch_bounds__(32)
void k_gemm(const bf16_t* __restrict__ X, const bf16_t* __restrict__ Wb,
            const float* __restrict__ bias, bf16_t* __restrict__ Y,
            float* __restrict__ stats, int M, int O, int Cp) {
  const int lane = threadIdx.x & 31;
  const int row  = lane & 15;
  const int half = lane >> 4;
  const int m0 = blockIdx.x * 16;
  const int n0 = blockIdx.y * (16 * NT);

  const bf16_t* arow = X + (size_t)(m0 + row) * Cp;
  const bf16_t* brow[NT];
#pragma unroll
  for (int t = 0; t < NT; ++t) brow[t] = Wb + (size_t)(n0 + t * 16 + row) * Cp;

  f32x8 acc[NT];
#pragma unroll
  for (int t = 0; t < NT; ++t) acc[t] = (f32x8){};

  for (int kb = 0; kb < Cp; kb += 32) {
    // A lane: row (lane&15); 16 consecutive K at kb + half*16
    bf16x8 a0 = *(const bf16x8*)(arow + kb + half * 16);
    bf16x8 a1 = *(const bf16x8*)(arow + kb + half * 16 + 8);
    bf16x16 a = __builtin_shufflevector(a0, a1, 0, 1, 2, 3, 4, 5, 6, 7,
                                        8, 9, 10, 11, 12, 13, 14, 15);
    if (kb + 32 < Cp) __builtin_prefetch(arow + kb + 32, 0, 1);  // global_prefetch_b8
#pragma unroll
    for (int t = 0; t < NT; ++t) {
      // B lane: column (lane&15); 16 consecutive K at kb + half*16
      bf16x16 b = *(const bf16x16*)(brow[t] + kb + half * 16);
      acc[t] = __builtin_amdgcn_wmma_f32_16x16x32_bf16(
          /*neg_a=*/false, a, /*neg_b=*/false, b,
          /*c_mod=*/(short)0, acc[t], /*reuse_a=*/false, /*reuse_b=*/false);
    }
  }

#pragma unroll
  for (int t = 0; t < NT; ++t) {
    const int n = n0 + t * 16 + row;  // column identical for all 8 elems of this lane
    const float bn = bias[n];
    float s = 0.f, sq = 0.f;
#pragma unroll
    for (int j = 0; j < 8; ++j) {
      float d = acc[t][j] + bn;  // C/D layout: VGPR j -> row m0 + j + 8*half
      s  += d;
      sq += d * d;
      Y[(size_t)(m0 + j + 8 * half) * O + n] = (bf16_t)d;
    }
    atomicAdd(&stats[n], s);
    atomicAdd(&stats[O + n], sq);
  }
}

// ---------------------------------------------------------------------------
// BatchNorm (global stats over M rows) + ReLU, in place on bf16 activations.
// ---------------------------------------------------------------------------
__global__ void k_bn_relu(bf16_t* __restrict__ Y, const float* __restrict__ stats,
                          float invM, int O, size_t total) {
  size_t i = (size_t)blockIdx.x * blockDim.x + threadIdx.x;
  if (i >= total) return;
  int o = (int)(i % O);
  float mean = stats[o] * invM;
  float var  = stats[O + o] * invM - mean * mean;
  float v = ((float)Y[i] - mean) * rsqrtf(var + BN_EPS);
  Y[i] = (bf16_t)fmaxf(v, 0.f);
}

// ---------------------------------------------------------------------------
// Max-pool over K samples, write f32 into concat destination.
// ---------------------------------------------------------------------------
__global__ void k_maxpool(const bf16_t* __restrict__ Xn, float* __restrict__ dst,
                          int BS, int K, int O, int Ctot, int choff) {
  int i = blockIdx.x * blockDim.x + threadIdx.x;
  if (i >= BS * O) return;
  int o  = i % O;
  int bs = i / O;
  const bf16_t* p = Xn + (size_t)bs * K * O + o;
  float m = -1e30f;
  for (int k = 0; k < K; ++k) m = fmaxf(m, (float)p[(size_t)k * O]);
  dst[(size_t)bs * Ctot + choff + o] = m;
}

// ---------------------------------------------------------------------------
// Host-side: one MLP stack (conv-weight, zero-stats, GEMM, BN+ReLU per layer)
// ---------------------------------------------------------------------------
static bf16_t* run_mlp(const float* const* IN, int base, int C0, int Cp0,
                       const int* och, int nL, int M,
                       bf16_t* bufA, bf16_t* bufB, bf16_t* wbuf, float* stats,
                       hipStream_t stream) {
  bf16_t* in = bufA;
  bf16_t* out = bufB;
  int C = C0, Cp = Cp0;
  for (int l = 0; l < nL; ++l) {
    const float* W = IN[base + 2 * l];
    const float* bv = IN[base + 2 * l + 1];
    int O = och[l];
    int wt = O * Cp;
    k_convw<<<(wt + 255) / 256, 256, 0, stream>>>(W, wbuf, C, Cp, wt);
    k_zero<<<(2 * O + 255) / 256, 256, 0, stream>>>(stats, 2 * O);
    if (O % 64 == 0) {
      dim3 g((unsigned)(M / 16), (unsigned)(O / 64));
      k_gemm<4><<<g, 32, 0, stream>>>(in, wbuf, bv, out, stats, M, O, Cp);
    } else {
      dim3 g((unsigned)(M / 16), (unsigned)(O / 32));
      k_gemm<2><<<g, 32, 0, stream>>>(in, wbuf, bv, out, stats, M, O, Cp);
    }
    size_t total = (size_t)M * O;
    k_bn_relu<<<(unsigned)((total + 255) / 256), 256, 0, stream>>>(
        out, stats, 1.0f / (float)M, O, total);
    bf16_t* t = in; in = out; out = t;
    C = O; Cp = O;
  }
  return in;  // last-written buffer
}

static inline size_t aln(size_t x) { return (x + 255) & ~(size_t)255; }

extern "C" void kernel_launch(void* const* d_in, const int* in_sizes, int n_in,
                              void* d_out, int out_size, void* d_ws, size_t ws_size,
                              hipStream_t stream) {
  (void)in_sizes; (void)n_in; (void)out_size; (void)ws_size;

  const float* IN[43];
  for (int i = 0; i < 43; ++i) IN[i] = (const float*)d_in[i];

  // ----- workspace carve -----
  char* ws = (char*)d_ws;
  size_t off = 0;
  size_t o_pts1 = off; off += aln((size_t)8 * 4096 * 3 * 4);
  size_t o_l1x  = off; off += aln((size_t)8 * 512 * 3 * 4);
  size_t o_l2x  = off; off += aln((size_t)8 * 128 * 3 * 4);
  size_t o_dist = off; off += aln((size_t)8 * 4096 * 4);
  size_t o_fidx = off; off += aln((size_t)8 * 512 * 4);
  size_t o_gidx = off; off += aln((size_t)8 * 512 * 128 * 4);
  size_t o_l1p  = off; off += aln((size_t)8 * 512 * 320 * 4);
  size_t o_l2p  = off; off += aln((size_t)8 * 128 * 640 * 4);
  size_t o_stat = off; off += aln((size_t)2048 * 4);
  size_t o_wbuf = off; off += aln((size_t)1024 * 672 * 2);
  size_t o_bufA = off; off += aln((size_t)524288 * 128 * 2);
  size_t o_bufB = off; off += aln((size_t)524288 * 128 * 2);

  float*  pts1   = (float*)(ws + o_pts1);
  float*  l1_xyz = (float*)(ws + o_l1x);
  float*  l2_xyz = (float*)(ws + o_l2x);
  float*  dist   = (float*)(ws + o_dist);
  int*    fidx   = (int*)(ws + o_fidx);
  int*    gidx   = (int*)(ws + o_gidx);
  float*  l1p    = (float*)(ws + o_l1p);
  float*  l2p    = (float*)(ws + o_l2p);
  float*  stats  = (float*)(ws + o_stat);
  bf16_t* wbuf   = (bf16_t*)(ws + o_wbuf);
  bf16_t* bufA   = (bf16_t*)(ws + o_bufA);
  bf16_t* bufB   = (bf16_t*)(ws + o_bufB);

  // ----- xyz (B,3,N) -> pts (B,N,3) -----
  k_transpose<<<(8 * 3 * 4096 + 255) / 256, 256, 0, stream>>>(IN[0], pts1, 8, 4096);

  // ===================== SA1 (MSG) =====================
  k_fps<<<8, 256, 0, stream>>>(pts1, dist, fidx, 4096, 512);
  k_gather3<<<(8 * 512 * 3 + 255) / 256, 256, 0, stream>>>(pts1, fidx, l1_xyz, 4096, 512,
                                                           8 * 512 * 3);
  {
    const float r2[3] = {0.01f, 0.04f, 0.16f};
    const int ks[3] = {16, 32, 128};
    const int och[3][3] = {{32, 32, 64}, {64, 64, 128}, {64, 96, 128}};
    int choff = 0;
    for (int sc = 0; sc < 3; ++sc) {
      int K = ks[sc];
      int rows = 8 * 512 * K;
      k_ball<<<(8 * 512 + 255) / 256, 256, 0, stream>>>(pts1, l1_xyz, gidx, 4096, 512, K,
                                                        r2[sc]);
      k_group_first<<<(rows + 255) / 256, 256, 0, stream>>>(pts1, l1_xyz, gidx, bufA, 4096,
                                                            512, K, 32, rows);
      bf16_t* res = run_mlp(IN, 1 + sc * 6, 3, 32, och[sc], 3, rows, bufA, bufB, wbuf,
                            stats, stream);
      int O = och[sc][2];
      k_maxpool<<<(8 * 512 * O + 255) / 256, 256, 0, stream>>>(res, l1p, 8 * 512, K, O,
                                                               320, choff);
      choff += O;
    }
  }

  // ===================== SA2 (MSG) =====================
  k_fps<<<8, 256, 0, stream>>>(l1_xyz, dist, fidx, 512, 128);
  k_gather3<<<(8 * 128 * 3 + 255) / 256, 256, 0, stream>>>(l1_xyz, fidx, l2_xyz, 512, 128,
                                                           8 * 128 * 3);
  {
    const float r2[3] = {0.04f, 0.16f, 0.64f};
    const int ks[3] = {32, 64, 128};
    const int och[3][3] = {{64, 64, 128}, {128, 128, 256}, {128, 128, 256}};
    int choff = 0;
    for (int sc = 0; sc < 3; ++sc) {
      int K = ks[sc];
      int rows = 8 * 128 * K;
      k_ball<<<(8 * 128 + 255) / 256, 256, 0, stream>>>(l1_xyz, l2_xyz, gidx, 512, 128, K,
                                                        r2[sc]);
      k_group_feat<<<(rows + 255) / 256, 256, 0, stream>>>(l1_xyz, l2_xyz, l1p, gidx, bufA,
                                                           512, 128, K, 320, 352, rows);
      bf16_t* res = run_mlp(IN, 19 + sc * 6, 323, 352, och[sc], 3, rows, bufA, bufB, wbuf,
                            stats, stream);
      int O = och[sc][2];
      k_maxpool<<<(8 * 128 * O + 255) / 256, 256, 0, stream>>>(res, l2p, 8 * 128, K, O,
                                                               640, choff);
      choff += O;
    }
  }

  // ===================== SA3 (group all) =====================
  {
    const int och[3] = {256, 512, 1024};
    int rows = 8 * 128;  // (B, S=1, K=128)
    k_group_all<<<(rows + 255) / 256, 256, 0, stream>>>(l2_xyz, l2p, bufA, 128, 640, 672,
                                                        rows);
    bf16_t* res = run_mlp(IN, 37, 643, 672, och, 3, rows, bufA, bufB, wbuf, stats, stream);
    k_maxpool<<<(8 * 1024 + 255) / 256, 256, 0, stream>>>(res, (float*)d_out, 8, 128, 1024,
                                                          1024, 0);
  }
}
